// SWA_33397665693797
// MI455X (gfx1250) — matrix operand order repaired and checked by reading, not compiled, not run on hardware
//
#include <hip/hip_runtime.h>
#include <stdint.h>

// ---------- types ----------
typedef __attribute__((ext_vector_type(16))) __bf16 v16bf;
typedef __attribute__((ext_vector_type(8)))  float  v8f;
typedef __attribute__((ext_vector_type(4)))  unsigned int u32x4;

#define TOKENS 100352   // B * 112 * 112
#define CCH    256
#define HWDIM  112
#define NSEQ   12544    // 112*112
#define NWIN   1792     // 12544 / 7
#define NBATCH 8

// ---------- TDM availability ----------
#if defined(__has_builtin)
#  if __has_builtin(__builtin_amdgcn_tensor_load_to_lds) && \
      __has_builtin(__builtin_amdgcn_s_wait_tensorcnt)
#    define HAVE_TDM 1
#  endif
#endif
#ifndef HAVE_TDM
#  define HAVE_TDM 0
#endif

#if HAVE_TDM
typedef __attribute__((ext_vector_type(4))) unsigned int tdm_u4;
typedef __attribute__((ext_vector_type(8))) int          tdm_i8;
typedef __attribute__((ext_vector_type(4))) int          tdm_i4;

// 1-D TDM copy: 4096 contiguous 2-byte elements (8 KB) global -> LDS.
// D# per CDNA5 ISA ch.8: group0 = {count=1 | lds_addr | global_addr | type=2},
// group1 = {data_size=1(2B), tensor_dim0=4096, tensor_dim1=1, tile_dim0=4096,
//           tile_dim1=1, tensor_dim0_stride=4096}. Groups 2/3 unused (<=2-D).
// This toolchain exposes the 6-arg builtin (extra int32x8 group + cpol).
__device__ inline void tdm_load_8kb(const uint16_t* gsrc, uint16_t* lds_dst) {
  uint64_t ga = (uint64_t)(uintptr_t)gsrc;
  uint32_t la = (uint32_t)(uintptr_t)(void*)lds_dst;   // low 32 bits = LDS offset
  tdm_u4 g0 = { 1u,                                   // count=1, user mode
                la,                                   // lds_addr
                (uint32_t)ga,                         // global_addr[31:0]
                ((uint32_t)(ga >> 32) & 0x01FFFFFFu)  // global_addr[56:32]
                    | (2u << 30) };                   // type=2 ("image")
  tdm_i8 g1 = { (int)(1u << 16),                      // data_size=1 -> 2 bytes
                (int)(4096u << 16),                   // tensor_dim0[15:0]=4096
                (int)(1u << 16),                      // dim0 hi=0, tensor_dim1=1
                (int)(4096u << 16),                   // dim1 hi=0, tile_dim0=4096
                1,                                    // tile_dim1=1
                4096,                                 // tensor_dim0_stride lo32
                0, 0 };
  tdm_i4 gz4 = { 0, 0, 0, 0 };
  tdm_i8 gz8 = { 0, 0, 0, 0, 0, 0, 0, 0 };
  __builtin_amdgcn_tensor_load_to_lds(g0, g1, gz4, gz4, gz8, 0);
}
#endif

// ---------- helpers ----------
__device__ inline uint16_t f2bf(float f) {
  uint32_t u = __float_as_uint(f);
  u += 0x7FFFu + ((u >> 16) & 1u);       // round-to-nearest-even
  return (uint16_t)(u >> 16);
}
__device__ inline float bf2f(uint16_t h) {
  return __uint_as_float(((uint32_t)h) << 16);
}

// A fragment 16x32 bf16: lane holds row=lane%16, contiguous 16 K values
__device__ inline v16bf load_frag_a(const uint16_t* p) {
  union { v16bf v; u32x4 q[2]; } r;
  r.q[0] = *reinterpret_cast<const u32x4*>(p);
  r.q[1] = *reinterpret_cast<const u32x4*>(p + 8);
  return r.v;
}
// B fragment 32x16 bf16: lane holds col=lane%16, contiguous 16 K values
__device__ inline v16bf load_frag_b(const uint16_t* p) {
  union { v16bf v; u32x4 q[2]; } r;
  r.q[0] = *reinterpret_cast<const u32x4*>(p);
  r.q[1] = *reinterpret_cast<const u32x4*>(p + 8);
  return r.v;
}

// block(256) sum: wave32 butterfly + 8-partial LDS combine (red[] >= 8 floats)
__device__ inline float blk_sum256(float v, float* red) {
  #pragma unroll
  for (int m = 16; m >= 1; m >>= 1) v += __shfl_xor(v, m);
  int tid = threadIdx.x;
  if ((tid & 31) == 0) red[tid >> 5] = v;
  __syncthreads();
  float t = 0.f;
  #pragma unroll
  for (int i = 0; i < 8; ++i) t += red[i];
  __syncthreads();
  return t;
}

// ---------- K0: weights fp32 -> bf16 ----------
__global__ __launch_bounds__(256) void k0_cvt(const float* w0, const float* w1,
                                              const float* w2, const float* w3,
                                              uint16_t* o0, uint16_t* o1,
                                              uint16_t* o2, uint16_t* o3) {
  int i = blockIdx.x * 256 + threadIdx.x;          // 0..65535
  int which = blockIdx.y;
  const float* s = which == 0 ? w0 : which == 1 ? w1 : which == 2 ? w2 : w3;
  uint16_t*    d = which == 0 ? o0 : which == 1 ? o1 : which == 2 ? o2 : o3;
  d[i] = f2bf(s[i]);
}

// ---------- K1: roll(+3,+3) + LayerNorm -> bf16 tokens (shifted order) ----------
__global__ __launch_bounds__(256) void k1_shift_ln(const float* __restrict__ x,
                                                   const float* __restrict__ g,
                                                   const float* __restrict__ b,
                                                   uint16_t* __restrict__ out) {
  __shared__ float red[8];
  int t = blockIdx.x, c = threadIdx.x;
  int bb = t / NSEQ, rem = t % NSEQ, h = rem / HWDIM, w = rem % HWDIM;
  int hs = h - 3; if (hs < 0) hs += HWDIM;
  int ws2 = w - 3; if (ws2 < 0) ws2 += HWDIM;
  float v = x[(((size_t)bb * HWDIM + hs) * HWDIM + ws2) * CCH + c];
  float mu  = blk_sum256(v, red) * (1.0f / CCH);
  float d0  = v - mu;
  float var = blk_sum256(d0 * d0, red) * (1.0f / CCH);
  float y = d0 * rsqrtf(var + 1e-5f) * g[c] + b[c];
  out[(size_t)t * CCH + c] = f2bf(y);
}

// ---------- shared A-tile staging for the GEMM kernels ----------
__device__ inline void stage_a_tile(const uint16_t* __restrict__ A, int row0,
                                    uint16_t* Alds, int tid) {
#if HAVE_TDM
  if (tid < 32) {                                  // one wave issues the DMA
    tdm_load_8kb(A + (size_t)row0 * CCH, Alds);
    __builtin_amdgcn_s_wait_tensorcnt(0);
  }
#else
  const u32x4* Ag = reinterpret_cast<const u32x4*>(A + (size_t)row0 * CCH);
  u32x4* Al = reinterpret_cast<u32x4*>(Alds);
  #pragma unroll
  for (int i = tid; i < 512; i += 256) Al[i] = Ag[i];
#endif
  __syncthreads();
}

// ---------- K2: fused QKV GEMM (WMMA bf16 -> f32) ----------
__global__ __launch_bounds__(256) void k2_qkv_gemm(const uint16_t* __restrict__ A,
    const uint16_t* __restrict__ Wq, const uint16_t* __restrict__ Wk,
    const uint16_t* __restrict__ Wv,
    const float* __restrict__ bq, const float* __restrict__ bk,
    const float* __restrict__ bv,
    uint16_t* __restrict__ Oq, uint16_t* __restrict__ Ok,
    uint16_t* __restrict__ Ov) {
  int which = blockIdx.y;
  const uint16_t* W    = which == 0 ? Wq : which == 1 ? Wk : Wv;
  const float*    bias = which == 0 ? bq : which == 1 ? bk : bv;
  uint16_t*       O    = which == 0 ? Oq : which == 1 ? Ok : Ov;
  float scale = which == 0 ? 0.0625f : 1.0f;       // q *= d^-0.5 = 1/16

  __shared__ uint16_t Alds[16 * CCH];              // 8 KB tile
  int tid = threadIdx.x;
  int row0 = blockIdx.x * 16;
  __builtin_prefetch(A + (size_t)(row0 + 16) * CCH, 0, 0);
  stage_a_tile(A, row0, Alds, tid);

  int wave = tid >> 5, lane = tid & 31;
  int hl = lane >> 4, l15 = lane & 15;
  int n0 = wave * 32;
  const uint16_t* Wr0 = W + (size_t)(n0 + l15) * CCH;
  const uint16_t* Wr1 = W + (size_t)(n0 + 16 + l15) * CCH;
  const uint16_t* Ar  = Alds + l15 * CCH;
  v8f acc0 = {}; v8f acc1 = {};
  #pragma unroll
  for (int kk = 0; kk < 8; ++kk) {
    int kbA = kk * 32 + hl * 16;
    int kbB = kk * 32 + hl * 16;
    v16bf af = load_frag_a(Ar + kbA);
    v16bf b0 = load_frag_b(Wr0 + kbB);
    v16bf b1 = load_frag_b(Wr1 + kbB);
    acc0 = __builtin_amdgcn_wmma_f32_16x16x32_bf16(false, af, false, b0,
                                                   (short)0, acc0, false, false);
    acc1 = __builtin_amdgcn_wmma_f32_16x16x32_bf16(false, af, false, b1,
                                                   (short)0, acc1, false, false);
  }
  #pragma unroll
  for (int i = 0; i < 8; ++i) {
    int m = hl * 8 + i;
    size_t base = (size_t)(row0 + m) * CCH;
    float v0 = (acc0[i] + bias[n0 + l15]) * scale;
    float v1 = (acc1[i] + bias[n0 + 16 + l15]) * scale;
    O[base + n0 + l15]      = f2bf(v0);
    O[base + n0 + 16 + l15] = f2bf(v1);
  }
}

// ---------- K3: local causal attention, window 7, look-back 1 ----------
__global__ __launch_bounds__(256) void k3_attn(const uint16_t* __restrict__ Q,
                                               const uint16_t* __restrict__ K,
                                               const uint16_t* __restrict__ V,
                                               uint16_t* __restrict__ O) {
  __shared__ float Qs[7 * CCH];
  __shared__ float Ks[14 * CCH];
  __shared__ float Vs[14 * CCH];
  __shared__ float simS[98];
  __shared__ float invn[14];
  int tid = threadIdx.x;
  int b = blockIdx.y, wi = blockIdx.x;             // wi in [0,1792)
  size_t base = (size_t)b * NSEQ + (size_t)wi * 7;

  for (int idx = tid; idx < 7 * CCH; idx += 256)
    Qs[idx] = bf2f(Q[(base + (idx >> 8)) * CCH + (idx & 255)]);
  for (int idx = tid; idx < 14 * CCH; idx += 256) {
    int j = idx >> 8, c = idx & 255;
    int jt = (j < 7) ? (wi > 0 ? (wi - 1) * 7 + j : -1) : wi * 7 + (j - 7);
    float kv = 0.f, vv = 0.f;
    if (jt >= 0) {
      size_t tok = (size_t)b * NSEQ + jt;
      kv = bf2f(K[tok * CCH + c]);
      vv = bf2f(V[tok * CCH + c]);
    }
    Ks[idx] = kv; Vs[idx] = vv;
  }
  __syncthreads();

  if (tid < 14) {                                  // shared-QK: L2 norm of keys
    float ss = 0.f;
    for (int c = 0; c < CCH; ++c) { float kv = Ks[tid * CCH + c]; ss += kv * kv; }
    invn[tid] = 1.0f / fmaxf(sqrtf(ss), 1e-12f);
  }
  __syncthreads();

  if (tid < 98) {
    int qi = tid / 14, j = tid % 14;
    float d = 0.f;
    for (int c = 0; c < CCH; ++c) d += Qs[qi * CCH + c] * Ks[j * CCH + c];
    d *= invn[j];
    int tq = wi * 7 + qi;
    int tk = (j < 7) ? (wi > 0 ? (wi - 1) * 7 + j : -1) : wi * 7 + (j - 7);
    if (tk == tq) d = -5e4f;                       // TOKEN_SELF_ATTN_VALUE
    if (tk < 0 || tk > tq) d = -3.402823466e38f;   // causal / pad mask
    simS[tid] = d;
  }
  __syncthreads();

  if (tid < 7) {
    float mx = -3.402823466e38f;
    for (int j = 0; j < 14; ++j) mx = fmaxf(mx, simS[tid * 14 + j]);
    float e[14], s = 0.f;
    for (int j = 0; j < 14; ++j) { e[j] = expf(simS[tid * 14 + j] - mx); s += e[j]; }
    float inv = 1.f / s;
    for (int j = 0; j < 14; ++j) simS[tid * 14 + j] = e[j] * inv;
  }
  __syncthreads();

  for (int idx = tid; idx < 7 * CCH; idx += 256) {
    int qi = idx >> 8, c = idx & 255;
    float acc = 0.f;
    #pragma unroll
    for (int j = 0; j < 14; ++j) acc += simS[qi * 14 + j] * Vs[j * CCH + c];
    O[(base + qi) * CCH + c] = f2bf(acc);
  }
}

// ---------- K4: un-roll + residual (fp32) + LayerNorm2 (bf16) ----------
__global__ __launch_bounds__(256) void k4_resid_ln(const uint16_t* __restrict__ attn,
                                                   const float* __restrict__ x,
                                                   const float* __restrict__ g,
                                                   const float* __restrict__ b,
                                                   float* __restrict__ resid,
                                                   uint16_t* __restrict__ ln2) {
  __shared__ float red[8];
  int t = blockIdx.x, c = threadIdx.x;
  int bb = t / NSEQ, rem = t % NSEQ, h = rem / HWDIM, w = rem % HWDIM;
  int hs = h + 3; if (hs >= HWDIM) hs -= HWDIM;
  int ws2 = w + 3; if (ws2 >= HWDIM) ws2 -= HWDIM;
  size_t tsh = (size_t)bb * NSEQ + (size_t)hs * HWDIM + ws2;
  float r = bf2f(attn[tsh * CCH + c]) + x[(size_t)t * CCH + c];
  resid[(size_t)t * CCH + c] = r;
  float mu  = blk_sum256(r, red) * (1.0f / CCH);
  float d0  = r - mu;
  float var = blk_sum256(d0 * d0, red) * (1.0f / CCH);
  float y = d0 * rsqrtf(var + 1e-5f) * g[c] + b[c];
  ln2[(size_t)t * CCH + c] = f2bf(y);
}

// ---------- K5: MLP GEMM (WMMA) + exact GELU + residual -> fp32 out ----------
__global__ __launch_bounds__(256) void k5_mlp_gemm(const uint16_t* __restrict__ A,
    const uint16_t* __restrict__ W, const float* __restrict__ bias,
    const float* __restrict__ resid, float* __restrict__ out) {
  __shared__ uint16_t Alds[16 * CCH];
  int tid = threadIdx.x;
  int row0 = blockIdx.x * 16;
  __builtin_prefetch(A + (size_t)(row0 + 16) * CCH, 0, 0);
  stage_a_tile(A, row0, Alds, tid);

  int wave = tid >> 5, lane = tid & 31;
  int hl = lane >> 4, l15 = lane & 15;
  int n0 = wave * 32;
  const uint16_t* Wr0 = W + (size_t)(n0 + l15) * CCH;
  const uint16_t* Wr1 = W + (size_t)(n0 + 16 + l15) * CCH;
  const uint16_t* Ar  = Alds + l15 * CCH;
  v8f acc0 = {}; v8f acc1 = {};
  #pragma unroll
  for (int kk = 0; kk < 8; ++kk) {
    int kbA = kk * 32 + hl * 16;
    int kbB = kk * 32 + hl * 16;
    v16bf af = load_frag_a(Ar + kbA);
    v16bf b0 = load_frag_b(Wr0 + kbB);
    v16bf b1 = load_frag_b(Wr1 + kbB);
    acc0 = __builtin_amdgcn_wmma_f32_16x16x32_bf16(false, af, false, b0,
                                                   (short)0, acc0, false, false);
    acc1 = __builtin_amdgcn_wmma_f32_16x16x32_bf16(false, af, false, b1,
                                                   (short)0, acc1, false, false);
  }
  #pragma unroll
  for (int i = 0; i < 8; ++i) {
    int m = hl * 8 + i;
    size_t base = (size_t)(row0 + m) * CCH;
    float v0 = acc0[i] + bias[n0 + l15];
    float v1 = acc1[i] + bias[n0 + 16 + l15];
    float g0 = 0.5f * v0 * (1.0f + erff(v0 * 0.70710678118654752f));
    float g1 = 0.5f * v1 * (1.0f + erff(v1 * 0.70710678118654752f));
    out[base + n0 + l15]      = g0 + resid[base + n0 + l15];
    out[base + n0 + 16 + l15] = g1 + resid[base + n0 + 16 + l15];
  }
}

// ---------- host launcher ----------
extern "C" void kernel_launch(void* const* d_in, const int* in_sizes, int n_in,
                              void* d_out, int out_size, void* d_ws, size_t ws_size,
                              hipStream_t stream) {
  const float* x     = (const float*)d_in[0];
  const float* ln_g  = (const float*)d_in[1];
  const float* ln_b  = (const float*)d_in[2];
  const float* wq_w  = (const float*)d_in[3];
  const float* wq_b  = (const float*)d_in[4];
  const float* wk_w  = (const float*)d_in[5];
  const float* wk_b  = (const float*)d_in[6];
  const float* wv_w  = (const float*)d_in[7];
  const float* wv_b  = (const float*)d_in[8];
  const float* mlp_w = (const float*)d_in[9];
  const float* mlp_b = (const float*)d_in[10];
  float* out = (float*)d_out;

  // workspace layout: 4 bf16 weight mats (512KB) + 4 bf16 token bufs (~206MB)
  const size_t WELEM = 256 * 256;          // 65536
  const size_t TELEM = (size_t)TOKENS * CCH;
  uint16_t* wq_bf  = (uint16_t*)d_ws;
  uint16_t* wk_bf  = wq_bf + WELEM;
  uint16_t* wv_bf  = wk_bf + WELEM;
  uint16_t* wm_bf  = wv_bf + WELEM;
  uint16_t* buf_ln = wm_bf + WELEM;        // LN1 out; later reused for attn out
  uint16_t* buf_q  = buf_ln + TELEM;
  uint16_t* buf_k  = buf_q + TELEM;
  uint16_t* buf_v  = buf_k + TELEM;        // later reused for LN2 out
  uint16_t* buf_attn = buf_ln;             // reuse: LN1 dead after K2
  float*    resid  = (float*)buf_q;        // reuse: q+k regions (fits fp32 exactly)
  uint16_t* buf_ln2 = buf_v;               // reuse: v dead after K3

  dim3 b256(256);
  k0_cvt<<<dim3(256, 4), b256, 0, stream>>>(wq_w, wk_w, wv_w, mlp_w,
                                            wq_bf, wk_bf, wv_bf, wm_bf);
  k1_shift_ln<<<dim3(TOKENS), b256, 0, stream>>>(x, ln_g, ln_b, buf_ln);
  k2_qkv_gemm<<<dim3(TOKENS / 16, 3), b256, 0, stream>>>(
      buf_ln, wq_bf, wk_bf, wv_bf, wq_b, wk_b, wv_b, buf_q, buf_k, buf_v);
  k3_attn<<<dim3(NWIN, NBATCH), b256, 0, stream>>>(buf_q, buf_k, buf_v, buf_attn);
  k4_resid_ln<<<dim3(TOKENS), b256, 0, stream>>>(buf_attn, x, ln_g, ln_b,
                                                 resid, buf_ln2);
  k5_mlp_gemm<<<dim3(TOKENS / 16), b256, 0, stream>>>(buf_ln2, wm_bf, mlp_b,
                                                      resid, out);
}
